// MMDLoss_15083925143681
// MI455X (gfx1250) — matrix operand order repaired and checked by reading, not compiled, not run on hardware
//
#include <hip/hip_runtime.h>
#include <hip/hip_bf16.h>
#include <stdint.h>

// ---------------------------------------------------------------------------
// MMD loss, fused Gram-tile kernel for gfx1250 (MI455X).
//   result = sum(exp(-a*||xi-xj||^2)) + sum(exp(-a*||yi-yj||^2))
//            - 2*sum(exp(-a*||xi-yj||^2))
// Strategy: bf16 hi/lo split (3-term product) + v_wmma_f32_16x16x32_bf16,
// fused exp + reduction so the 3x256MB Gram matrices never touch memory.
// ---------------------------------------------------------------------------

#define MMD_N      8192
#define MMD_D      64
#define MMD_ALPHA  0.1f
#define MMD_TILES  (MMD_N / 16)   // 512 tiles per dimension
#define MMD_WAVES  8              // waves per block (256 threads, wave32)

typedef __attribute__((ext_vector_type(16))) __bf16       v16bf;
typedef __attribute__((ext_vector_type(8)))  float        v8f;
typedef __attribute__((ext_vector_type(4)))  float        v4f;
typedef __attribute__((ext_vector_type(4)))  unsigned int u32x4;

// ---- bf16 helpers (manual RNE, no reliance on __bf16 arithmetic) ----------
__device__ __forceinline__ unsigned short f2bf(float f) {
    unsigned int u = __float_as_uint(f);
    u += 0x7FFFu + ((u >> 16) & 1u);       // round-to-nearest-even
    return (unsigned short)(u >> 16);
}
__device__ __forceinline__ float bf2f(unsigned short h) {
    return __uint_as_float(((unsigned int)h) << 16);
}

// ---- Kernel 1: row squared-norms + fp32 -> bf16 hi/lo split ---------------
__global__ __launch_bounds__(256) void mmd_prep(
    const float* __restrict__ x, const float* __restrict__ y,
    unsigned short* __restrict__ xhi, unsigned short* __restrict__ xlo,
    unsigned short* __restrict__ yhi, unsigned short* __restrict__ ylo,
    float* __restrict__ rx, float* __restrict__ ry)
{
    const int i = blockIdx.x * blockDim.x + threadIdx.x;   // row 0..8191
    float sx = 0.f, sy = 0.f;
    #pragma unroll 8
    for (int k = 0; k < MMD_D; ++k) {
        const int idx = i * MMD_D + k;
        float vx = x[idx];
        sx += vx * vx;
        unsigned short hx = f2bf(vx);
        xhi[idx] = hx;
        xlo[idx] = f2bf(vx - bf2f(hx));
        float vy = y[idx];
        sy += vy * vy;
        unsigned short hy = f2bf(vy);
        yhi[idx] = hy;
        ylo[idx] = f2bf(vy - bf2f(hy));
    }
    rx[i] = sx;
    ry[i] = sy;
}

// ---- WMMA fragment loaders (ISA 7.12.2 layouts, bf16 16x16x32) ------------
// A (16x32, MxK): lane L -> row M=L&15; K in [kb..kb+7] (V0-3) and
// [kb+16..kb+23] (V4-7), kb = k0 + 8*(L>=16). Row-major: two 16B loads.
__device__ __forceinline__ v16bf load_fragA(const unsigned short* __restrict__ m,
                                            int row0, int k0, int lane)
{
    const int r  = row0 + (lane & 15);
    const int kb = k0 + ((lane >> 4) << 3);
    const unsigned short* p = m + r * MMD_D + kb;
    union { v16bf v; u32x4 q[2]; } f;
    f.q[0] = *(const u32x4*)(p);
    f.q[1] = *(const u32x4*)(p + 16);
    return f.v;
}

// B (32x16, KxN): lane L -> col N=L&15; K in [kb..kb+7] (V0-3) and
// [kb+16..kb+23] (V4-7), kb = k0 + 8*(L>=16). B[k][n] = X[j0+n][k] -> row-major
// contiguous: two 16B loads from the untransposed array.
__device__ __forceinline__ v16bf load_fragB(const unsigned short* __restrict__ m,
                                            int col0, int k0, int lane)
{
    const int n  = col0 + (lane & 15);
    const int kb = k0 + ((lane >> 4) << 3);
    const unsigned short* p = m + n * MMD_D + kb;
    union { v16bf v; u32x4 q[2]; } f;
    f.q[0] = *(const u32x4*)(p);
    f.q[1] = *(const u32x4*)(p + 16);
    return f.v;
}

__device__ __forceinline__ v8f bfwmma(v16bf a, v16bf b, v8f c) {
    // (neg_a, A, neg_b, B, c_mod, C, reuse_a, reuse_b)
    return __builtin_amdgcn_wmma_f32_16x16x32_bf16(false, a, false, b,
                                                   (short)0, c, false, false);
}

// ---- Kernel 2: fused Gram tiles + exp + partial reduction -----------------
__global__ __launch_bounds__(256) void mmd_tiles(
    const unsigned short* __restrict__ xhi, const unsigned short* __restrict__ xlo,
    const unsigned short* __restrict__ yhi, const unsigned short* __restrict__ ylo,
    const float* __restrict__ rx, const float* __restrict__ ry,
    float* __restrict__ partials)
{
    const int lane = threadIdx.x & 31;
    const int wave = threadIdx.x >> 5;
    const int i0   = blockIdx.x << 4;          // this block's i-tile

    // Squared norms for the 8 rows this lane owns in the C/D layout:
    // lane L, VGPR r -> element (M = r + 8*(L>=16), N = L&15).
    const int hb = (lane >> 4) << 3;
    v4f ra0 = *(const v4f*)(rx + i0 + hb);
    v4f ra1 = *(const v4f*)(rx + i0 + hb + 4);
    v4f rb0 = *(const v4f*)(ry + i0 + hb);
    v4f rb1 = *(const v4f*)(ry + i0 + hb + 4);
    float rxi[8], ryi[8];
    #pragma unroll
    for (int r = 0; r < 4; ++r) {
        rxi[r] = ra0[r]; rxi[r + 4] = ra1[r];
        ryi[r] = rb0[r]; ryi[r + 4] = rb1[r];
    }

    // A-side fragments for this i-tile (reused across all 64 j-tiles).
    const v16bf axh0 = load_fragA(xhi, i0,  0, lane);
    const v16bf axh1 = load_fragA(xhi, i0, 32, lane);
    const v16bf axl0 = load_fragA(xlo, i0,  0, lane);
    const v16bf axl1 = load_fragA(xlo, i0, 32, lane);
    const v16bf ayh0 = load_fragA(yhi, i0,  0, lane);
    const v16bf ayh1 = load_fragA(yhi, i0, 32, lane);
    const v16bf ayl0 = load_fragA(ylo, i0,  0, lane);
    const v16bf ayl1 = load_fragA(ylo, i0, 32, lane);

    float lsum = 0.f;

    for (int jt = wave; jt < MMD_TILES; jt += MMD_WAVES) {
        const int j0 = jt << 4;
        v8f xx = {}, yy = {}, xy = {};

        {   // K chunk 0..31
            v16bf bxh = load_fragB(xhi, j0, 0, lane);
            v16bf bxl = load_fragB(xlo, j0, 0, lane);
            v16bf byh = load_fragB(yhi, j0, 0, lane);
            v16bf byl = load_fragB(ylo, j0, 0, lane);
            xx = bfwmma(axh0, bxh, xx); xx = bfwmma(axh0, bxl, xx); xx = bfwmma(axl0, bxh, xx);
            yy = bfwmma(ayh0, byh, yy); yy = bfwmma(ayh0, byl, yy); yy = bfwmma(ayl0, byh, yy);
            xy = bfwmma(axh0, byh, xy); xy = bfwmma(axh0, byl, xy); xy = bfwmma(axl0, byh, xy);
        }
        {   // K chunk 32..63
            v16bf bxh = load_fragB(xhi, j0, 32, lane);
            v16bf bxl = load_fragB(xlo, j0, 32, lane);
            v16bf byh = load_fragB(yhi, j0, 32, lane);
            v16bf byl = load_fragB(ylo, j0, 32, lane);
            xx = bfwmma(axh1, bxh, xx); xx = bfwmma(axh1, bxl, xx); xx = bfwmma(axl1, bxh, xx);
            yy = bfwmma(ayh1, byh, yy); yy = bfwmma(ayh1, byl, yy); yy = bfwmma(ayl1, byh, yy);
            xy = bfwmma(axh1, byh, xy); xy = bfwmma(axh1, byl, xy); xy = bfwmma(axl1, byh, xy);
        }

        const float rxj = rx[j0 + (lane & 15)];
        const float ryj = ry[j0 + (lane & 15)];
        #pragma unroll
        for (int r = 0; r < 8; ++r) {
            float dxx = rxi[r] + rxj - 2.0f * xx[r];
            float dyy = ryi[r] + ryj - 2.0f * yy[r];
            float dxy = rxi[r] + ryj - 2.0f * xy[r];
            lsum += __expf(-MMD_ALPHA * dxx) + __expf(-MMD_ALPHA * dyy)
                  - 2.0f * __expf(-MMD_ALPHA * dxy);
        }
    }

    // Deterministic wave32 reduction; one partial per wave (no float atomics).
    #pragma unroll
    for (int off = 16; off > 0; off >>= 1)
        lsum += __shfl_xor(lsum, off, 32);
    if (lane == 0)
        partials[blockIdx.x * MMD_WAVES + wave] = lsum;
}

// ---- Kernel 3: fixed-order final reduction --------------------------------
__global__ __launch_bounds__(256) void mmd_reduce(const float* __restrict__ partials,
                                                  float* __restrict__ out)
{
    __shared__ float s[256];
    const int t = threadIdx.x;
    float a = 0.f;
    for (int k = t; k < MMD_TILES * MMD_WAVES; k += 256)
        a += partials[k];
    s[t] = a;
    __syncthreads();
    for (int w = 128; w > 0; w >>= 1) {
        if (t < w) s[t] += s[t + w];
        __syncthreads();
    }
    if (t == 0) out[0] = s[0];
}

// ---------------------------------------------------------------------------
extern "C" void kernel_launch(void* const* d_in, const int* in_sizes, int n_in,
                              void* d_out, int out_size, void* d_ws, size_t ws_size,
                              hipStream_t stream)
{
    (void)in_sizes; (void)n_in; (void)out_size; (void)ws_size;
    const float* x = (const float*)d_in[0];
    const float* y = (const float*)d_in[1];

    // Workspace layout (~4.15 MB total)
    const size_t szBF = (size_t)MMD_N * MMD_D * sizeof(unsigned short); // 1 MB
    char* w = (char*)d_ws;
    unsigned short* xhi = (unsigned short*)(w);
    unsigned short* xlo = (unsigned short*)(w + 1 * szBF);
    unsigned short* yhi = (unsigned short*)(w + 2 * szBF);
    unsigned short* ylo = (unsigned short*)(w + 3 * szBF);
    float* rx       = (float*)(w + 4 * szBF);
    float* ry       = rx + MMD_N;
    float* partials = ry + MMD_N;   // 4096 floats

    mmd_prep<<<MMD_N / 256, 256, 0, stream>>>(x, y, xhi, xlo, yhi, ylo, rx, ry);
    mmd_tiles<<<MMD_TILES, 256, 0, stream>>>(xhi, xlo, yhi, ylo, rx, ry, partials);
    mmd_reduce<<<1, 256, 0, stream>>>(partials, (float*)d_out);
}
